// GATOccupancyPredictor_49022756716782
// MI455X (gfx1250) — matrix operand order repaired and checked by reading, not compiled, not run on hardware
//
#include <hip/hip_runtime.h>
#include <hip/hip_bf16.h>

// ---------------- problem constants ----------------
#define NPTS   3000
#define MPTS   3000
#define KN     6000          // total nodes
#define KP     6016          // padded to multiple of 32 (no tail in j loop)
#define HEADS  4
#define HID    64
#define FEAT   256           // HEADS*HID
#define R2     0.0025f       // 0.05^2
#define NEGS   0.2f

typedef __attribute__((ext_vector_type(16))) _Float16 v16h;
typedef __attribute__((ext_vector_type(8)))  _Float16 v8h;
typedef __attribute__((ext_vector_type(8)))  float    v8f;

// ---------------- kernel 1: build pts+|pts|^2 ----------------
__global__ void prep_pts_kernel(const float* __restrict__ pos,
                                const float* __restrict__ pnm,
                                float4* __restrict__ ptsq) {
  int j = blockIdx.x * 256 + threadIdx.x;
  if (j >= KP) return;
  float x = 0.f, y = 0.f, z = 0.f;
  if (j < NPTS) {
    x = pos[j]; y = pos[NPTS + j]; z = pos[2 * NPTS + j];
  } else if (j < KN) {
    int m = j - NPTS;
    x = pnm[m]; y = pnm[MPTS + m]; z = pnm[2 * MPTS + m];
  }
  float4 p; p.x = x; p.y = y; p.z = z; p.w = x*x + y*y + z*z;
  ptsq[j] = p;
}

// ---------------- kernel 2: h1 = pts @ W1, stored feature-major f16 ----------------
__global__ void gemm1_kernel(const float4* __restrict__ ptsq,
                             const float* __restrict__ W1,   // [3][256]
                             _Float16* __restrict__ hT) {    // [256][KP]
  int j = blockIdx.x;        // node
  int o = threadIdx.x;       // feature
  float v = 0.f;
  if (j < KN) {
    float4 p = ptsq[j];
    v = p.x * W1[o] + p.y * W1[256 + o] + p.z * W1[512 + o];
  }
  hT[(size_t)o * KP + j] = (_Float16)v;
}

// ---------------- kernel: h = x @ W (256x256), stored feature-major f16 ----------------
__global__ void gemm2_kernel(const float* __restrict__ x,    // [KP][256]
                             const float* __restrict__ W,    // [256][256]
                             _Float16* __restrict__ hT) {    // [256][KP]
  int j = blockIdx.x;
  int o = threadIdx.x;
  float v = 0.f;
  if (j < KN) {
    const float* xr = x + (size_t)j * FEAT;  // uniform per block -> scalar loads
    #pragma unroll 8
    for (int k = 0; k < FEAT; ++k) v += xr[k] * W[k * FEAT + o];
  }
  hT[(size_t)o * KP + j] = (_Float16)v;
}

// ---------------- kernel: attention score projections es/ed ----------------
__global__ void scores_kernel(const _Float16* __restrict__ hT,  // [256][KP]
                              const float* __restrict__ a_src,  // [4][64]
                              const float* __restrict__ a_dst,  // [4][64]
                              float* __restrict__ es,           // [KP][4]
                              float* __restrict__ ed) {
  int j  = blockIdx.x * 256 + threadIdx.x;
  int hd = blockIdx.y;
  if (j >= KP) return;
  const _Float16* hc = hT + (size_t)(hd * HID) * KP + j;
  float s = 0.f, d = 0.f;
  #pragma unroll 8
  for (int c = 0; c < HID; ++c) {
    float hv = (float)hc[(size_t)c * KP];
    s += hv * a_src[hd * HID + c];
    d += hv * a_dst[hd * HID + c];
  }
  es[j * HEADS + hd] = s;
  ed[j * HEADS + hd] = d;
}

// ---------------- the WMMA attention kernel (one GAT layer) ----------------
// block = 128 threads = 4 waves; wave w handles head w for a shared 16-row tile.
// out_i = relu( (sum_j w_ij h_j) / (sum_j w_ij) + bias ), w_ij = exp(leaky(ed_i+es_j))*mask
__global__ __launch_bounds__(128) void attn_kernel(
    const float4*  __restrict__ ptsq,   // [KP]
    const _Float16* __restrict__ hT,    // [256][KP] feature-major f16
    const float*   __restrict__ es,     // [KP][4]
    const float*   __restrict__ ed,     // [KP][4]
    const float*   __restrict__ bias,   // [256]
    float*         __restrict__ xout) { // [KP][256], rows < KN written
  __shared__ float4 pi_s[16];
  __shared__ __align__(16) float ed_s[HEADS][16];  // per-head contiguous -> b128 merges
  __shared__ __align__(16) _Float16 wt[HEADS][16][32];

  const int tid  = threadIdx.x;
  const int lane = tid & 31;
  const int head = tid >> 5;
  const int i0   = blockIdx.x * 16;

  if (tid < 16) pi_s[tid] = ptsq[i0 + tid];
  if (tid < 64) { int hd = tid >> 4, r = tid & 15; ed_s[hd][r] = ed[(i0 + r) * HEADS + hd]; }
  __syncthreads();

  v8f acc[4] = {v8f{}, v8f{}, v8f{}, v8f{}};
  v8f accS   = v8f{};
  v16h ones;
  #pragma unroll
  for (int i = 0; i < 16; ++i) ones[i] = (_Float16)1.0f;

  const int rr = lane & 15;       // row within tile (A) / column within feature tile (B)
  const int hi = lane >> 4;       // lane-half selector for K split
  const _Float16* hTh = hT + (size_t)(head * HID) * KP;

  for (int c0 = 0; c0 < KP; c0 += 32) {
    const int j = c0 + lane;
    const float4 pj  = ptsq[j];
    const float  esj = es[j * HEADS + head];
    const float  vmask = (j < KN) ? 1.0f : 0.0f;

    // prefetch next chunk's B columns (speculative)
    __builtin_prefetch(hTh + (size_t)rr * KP + (c0 + 32), 0, 3);

    // compute 16x32 w tile; lane owns column j, iterate rows r (wave-uniform LDS reads).
    // fully branchless: exp computed unconditionally, mask applied by multiply.
    #pragma unroll
    for (int r = 0; r < 16; ++r) {
      float4 pi = pi_s[r];
      float d2 = pi.w + pj.w - 2.0f * (pi.x * pj.x + pi.y * pj.y + pi.z * pj.z);
      float e  = ed_s[head][r] + esj;
      e = (e >= 0.f) ? e : NEGS * e;
      float m  = (d2 < R2) ? vmask : 0.0f;
      float w  = __expf(e) * m;
      wt[head][r][lane] = (_Float16)w;
    }
    // wave-private tile; DS ops are in-order within a wave -- just stop compiler reordering
    asm volatile("" ::: "memory");

    // A fragment (16x32 f16): halves 0..7 = K (hi?8..15:0..7), halves 8..15 = K 16+...
    v8h alo = *(const v8h*)&wt[head][rr][hi * 8];
    v8h ahi = *(const v8h*)&wt[head][rr][16 + hi * 8];
    v16h A = __builtin_shufflevector(alo, ahi, 0,1,2,3,4,5,6,7,8,9,10,11,12,13,14,15);
    asm volatile("" ::: "memory");

    // B fragments straight from feature-major global memory (two 16B runs per lane: K hi*8.. and 16+hi*8..)
    #pragma unroll
    for (int f = 0; f < 4; ++f) {
      const _Float16* p = hTh + (size_t)(f * 16 + rr) * KP + (c0 + hi * 8);
      v8h blo = *(const v8h*)p;
      v8h bhi = *(const v8h*)(p + 16);
      v16h Bf = __builtin_shufflevector(blo, bhi, 0,1,2,3,4,5,6,7,8,9,10,11,12,13,14,15);
      acc[f] = __builtin_amdgcn_wmma_f32_16x16x32_f16(false, A, false, Bf,
                                                      (short)0, acc[f], false, false);
    }
    // row-sum via WMMA against all-ones B (denominator of softmax)
    accS = __builtin_amdgcn_wmma_f32_16x16x32_f16(false, A, false, ones,
                                                  (short)0, accS, false, false);
  }

  // D layout: VGPR v, lanes 0-15: row v, col lane; lanes 16-31: row v+8, col lane-16
  #pragma unroll
  for (int v = 0; v < 8; ++v) {
    int row = i0 + v + hi * 8;
    // raw hardware reciprocal (v_rcp_f32) -- softmax normalizer, precision not critical
    float inv = __builtin_amdgcn_rcpf(accS[v] + 1e-20f);  // diag edge guarantees accS>0
    #pragma unroll
    for (int f = 0; f < 4; ++f) {
      int col = head * HID + f * 16 + rr;
      float val = acc[f][v] * inv + bias[col];
      xout[(size_t)row * FEAT + col] = fmaxf(val, 0.0f);
    }
  }
}

// ---------------- final fc: logits ----------------
__global__ void fc_kernel(const float* __restrict__ x,     // [KP][256]
                          const float* __restrict__ fw,    // [256][2]
                          const float* __restrict__ fb,    // [2]
                          float* __restrict__ out) {       // 6000 floats
  int f = blockIdx.x * 256 + threadIdx.x;
  if (f >= 6000) return;
  int node = MPTS + (f >> 1);
  int ch   = f & 1;
  const float* xr = x + (size_t)node * FEAT;
  float v = fb[ch];
  #pragma unroll 8
  for (int k = 0; k < FEAT; ++k) v += xr[k] * fw[k * 2 + ch];
  out[f] = v;
}

// ---------------- launcher ----------------
extern "C" void kernel_launch(void* const* d_in, const int* in_sizes, int n_in,
                              void* d_out, int out_size, void* d_ws, size_t ws_size,
                              hipStream_t stream) {
  const float* pos   = (const float*)d_in[0];
  const float* pnm   = (const float*)d_in[1];
  const float* W1    = (const float*)d_in[2];
  const float* as1   = (const float*)d_in[3];
  const float* ad1   = (const float*)d_in[4];
  const float* b1    = (const float*)d_in[5];
  const float* W2    = (const float*)d_in[6];
  const float* as2   = (const float*)d_in[7];
  const float* ad2   = (const float*)d_in[8];
  const float* b2    = (const float*)d_in[9];
  const float* fcw   = (const float*)d_in[10];
  const float* fcb   = (const float*)d_in[11];
  float* out = (float*)d_out;

  // workspace carve-up (256B aligned)
  char* ws = (char*)d_ws;
  constexpr size_t OFF_PTSQ = 0;                            // KP*16      =   96,256
  constexpr size_t OFF_HT   = 98304;                        // 256*KP*2   = 3,080,192
  constexpr size_t OFF_ES   = OFF_HT + 3080192;             // KP*4*4
  constexpr size_t OFF_ED   = OFF_ES + 98304;
  constexpr size_t OFF_X    = OFF_ED + 98304;               // KP*256*4   = 6,160,384
  float4*   ptsq = (float4*)(ws + OFF_PTSQ);
  _Float16* hT   = (_Float16*)(ws + OFF_HT);
  float*    es   = (float*)(ws + OFF_ES);
  float*    ed   = (float*)(ws + OFF_ED);
  float*    x    = (float*)(ws + OFF_X);

  dim3 b256(256);
  dim3 gKP((KP + 255) / 256);
  dim3 gScores((KP + 255) / 256, HEADS);

  // geometry
  prep_pts_kernel<<<gKP, b256, 0, stream>>>(pos, pnm, ptsq);

  // ---- GAT layer 1 ----
  gemm1_kernel<<<dim3(KP), b256, 0, stream>>>(ptsq, W1, hT);
  scores_kernel<<<gScores, b256, 0, stream>>>(hT, as1, ad1, es, ed);
  attn_kernel<<<dim3(KN / 16), dim3(128), 0, stream>>>(ptsq, hT, es, ed, b1, x);

  // ---- GAT layer 2 ----
  gemm2_kernel<<<dim3(KP), b256, 0, stream>>>(x, W2, hT);
  scores_kernel<<<gScores, b256, 0, stream>>>(hT, as2, ad2, es, ed);
  attn_kernel<<<dim3(KN / 16), dim3(128), 0, stream>>>(ptsq, hT, es, ed, b2, x);

  // ---- fc + slice/reshape ----
  fc_kernel<<<dim3((6000 + 255) / 256), b256, 0, stream>>>(x, fcw, fcb, out);
}
